// MultiheadLatentAttention_54520314855558
// MI455X (gfx1250) — matrix operand order repaired and checked by reading, not compiled, not run on hardware
//
#include <hip/hip_runtime.h>

typedef __attribute__((ext_vector_type(16))) _Float16 v16h;
typedef __attribute__((ext_vector_type(8)))  _Float16 v8h;
typedef __attribute__((ext_vector_type(8)))  float    v8f;

#define WMMA16(a, b, c) __builtin_amdgcn_wmma_f32_16x16x32_f16( \
    false, (a), false, (b), (short)0, (c), false, false)

__device__ inline v8f vzero8f() {
  v8f z = {0.f, 0.f, 0.f, 0.f, 0.f, 0.f, 0.f, 0.f};
  return z;
}

// ---- gfx1250 async global->LDS copy (ASYNCcnt-tracked), 16B per lane -------
__device__ inline unsigned lds_off32(const void* p) {
  return (unsigned)(unsigned long long)p;  // low 32 bits of flat addr = LDS offset
}
__device__ inline void async_copy_b128(unsigned lds_dst, const void* gsrc) {
  asm volatile("global_load_async_to_lds_b128 %0, %1, off"
               :: "v"(lds_dst), "v"(gsrc) : "memory");
}
__device__ inline void wait_async0() {
  asm volatile("s_wait_asynccnt 0x0" ::: "memory");
}

// A-fragment (16x32 f16, M x K) from row-major LDS tile.
__device__ inline v16h load_frag_a(const _Float16* smem, int stride, int rbase,
                                   int kbase, int lane) {
  int r  = rbase + (lane & 15);
  int k0 = kbase + ((lane >> 4) << 3);
  const _Float16* p = smem + r * stride + k0;
  v8h lo = *(const v8h*)(p);
  v8h hi = *(const v8h*)(p + 16);
  return __builtin_shufflevector(lo, hi, 0, 1, 2, 3, 4, 5, 6, 7,
                                 8, 9, 10, 11, 12, 13, 14, 15);
}

// B-fragment (32x16 f16, K x N) from K-contiguous ("[N][K]") LDS tile.
__device__ inline v16h load_frag_b(const _Float16* smem, int stride, int nbase,
                                   int kbase, int lane) {
  int n  = nbase + (lane & 15);
  int k0 = kbase + ((lane >> 4) << 3);
  const _Float16* p = smem + n * stride + k0;
  v8h lo = *(const v8h*)(p);
  v8h hi = *(const v8h*)(p + 16);
  return __builtin_shufflevector(lo, hi, 0, 1, 2, 3, 4, 5, 6, 7,
                                 8, 9, 10, 11, 12, 13, 14, 15);
}

// ---------------------------------------------------------------------------
// f32 -> f16 elementwise convert
// ---------------------------------------------------------------------------
__global__ __launch_bounds__(256) void cvt_f16(const float* __restrict__ in,
                                               _Float16* __restrict__ out,
                                               long n) {
  long i = (long)blockIdx.x * 256 + threadIdx.x;
  long stride = (long)gridDim.x * 256;
  for (; i < n; i += stride) out[i] = (_Float16)in[i];
}

// f32 [K,N] -> f16 [N,K] transpose-convert (weights; done once).
// block = 256 threads mapped 32 (n) x 8 (k); grid = (N/32, K/8).
__global__ __launch_bounds__(256) void cvt_f16_t(const float* __restrict__ in,
                                                 _Float16* __restrict__ out,
                                                 int K, int N) {
  int n = blockIdx.x * 32 + (threadIdx.x & 31);
  int k = blockIdx.y * 8 + (threadIdx.x >> 5);
  out[(size_t)n * K + k] = (_Float16)in[(size_t)k * N + n];
}

// ---------------------------------------------------------------------------
// GEMM: C[M,N] = A[M,K] @ Bt[N,K]^T ; A,Bt f16 global, C f16 or f32.
// Block 256 = 8 waves. Tile BM=128, BN=64, BK=32; double-buffered LDS fed by
// global_load_async_to_lds_b128. Requires M%128==0, N%64==0, K%32==0.
// ---------------------------------------------------------------------------
template <bool OUTF16>
__global__ __launch_bounds__(256) void gemm_f16_wmma(
    const _Float16* __restrict__ A, const _Float16* __restrict__ Bt,
    void* __restrict__ Cv, int M, int N, int K) {
  const int AS = 40;  // 32 + 8 pad halves; rows stay 16B aligned (80B)
  __shared__ __align__(16) _Float16 As[2][128 * 40];
  __shared__ __align__(16) _Float16 Bs[2][64 * 40];

  const int tid  = threadIdx.x;
  const int lane = tid & 31;
  const int wave = tid >> 5;
  const int wm   = (wave >> 1) * 32;
  const int wn   = (wave & 1) * 32;
  const int bm   = blockIdx.y * 128;
  const int bn   = blockIdx.x * 64;
  const int nk   = K >> 5;

  auto issue = [&](int kt, int buf) {
    // A tile: 128 rows x 32 halves = 512 x 16B chunks; 2 per thread
#pragma unroll
    for (int i = 0; i < 2; ++i) {
      int c = tid + i * 256;
      int r = c >> 2, seg = c & 3;
      const _Float16* src = A + (size_t)(bm + r) * K + kt * 32 + seg * 8;
      async_copy_b128(lds_off32(&As[buf][r * AS + seg * 8]), src);
    }
    // B tile: 64 rows x 32 halves = 256 x 16B chunks; 1 per thread
    {
      int r = tid >> 2, seg = tid & 3;
      const _Float16* src = Bt + (size_t)(bn + r) * K + kt * 32 + seg * 8;
      async_copy_b128(lds_off32(&Bs[buf][r * AS + seg * 8]), src);
    }
  };

  v8f acc00 = vzero8f(), acc01 = vzero8f(), acc10 = vzero8f(), acc11 = vzero8f();

  issue(0, 0);
  for (int kt = 0; kt < nk; ++kt) {
    const int cur = kt & 1;
    wait_async0();      // this wave's async copies of tile kt complete
    __syncthreads();    // everyone's tile kt resident; buffer cur^1 free
    if (kt + 1 < nk) issue(kt + 1, cur ^ 1);

    v16h a0 = load_frag_a(As[cur], AS, wm + 0, 0, lane);
    v16h a1 = load_frag_a(As[cur], AS, wm + 16, 0, lane);
    v16h b0 = load_frag_b(Bs[cur], AS, wn + 0, 0, lane);
    v16h b1 = load_frag_b(Bs[cur], AS, wn + 16, 0, lane);
    acc00 = WMMA16(a0, b0, acc00);
    acc01 = WMMA16(a0, b1, acc01);
    acc10 = WMMA16(a1, b0, acc10);
    acc11 = WMMA16(a1, b1, acc11);
  }

  // C/D layout: VGPR v -> row v + 8*(lane>=16); col = lane&15
  const int hs = (lane >> 4) * 8;
  const int nn = lane & 15;
#pragma unroll
  for (int v = 0; v < 8; ++v) {
    size_t r0 = (size_t)(bm + wm + v + hs);
    size_t c0 = (size_t)(bn + wn + nn);
    if (OUTF16) {
      _Float16* C = (_Float16*)Cv;
      C[r0 * N + c0]             = (_Float16)acc00[v];
      C[r0 * N + c0 + 16]        = (_Float16)acc01[v];
      C[(r0 + 16) * N + c0]      = (_Float16)acc10[v];
      C[(r0 + 16) * N + c0 + 16] = (_Float16)acc11[v];
    } else {
      float* C = (float*)Cv;
      C[r0 * N + c0]             = acc00[v];
      C[r0 * N + c0 + 16]        = acc01[v];
      C[(r0 + 16) * N + c0]      = acc10[v];
      C[(r0 + 16) * N + c0 + 16] = acc11[v];
    }
  }
}

// ---------------------------------------------------------------------------
// RoPE + head assembly (all f16 I/O): qhat scaled by 1/sqrt(HD), [B,H,S,HD].
// ---------------------------------------------------------------------------
__global__ __launch_bounds__(128) void mla_assemble(
    const _Float16* __restrict__ q_cont, const _Float16* __restrict__ q_rot,
    const _Float16* __restrict__ k_cont, const _Float16* __restrict__ k_rot,
    const _Float16* __restrict__ v_full, _Float16* __restrict__ qh,
    _Float16* __restrict__ kh, _Float16* __restrict__ vh, int S) {
  const int r = blockIdx.x;      // (b*S + s)*H + h
  const int d = threadIdx.x;     // 0..127
  const int h = r & 15;
  const int s = (r >> 4) % S;
  const int b = r / (16 * S);
  const size_t incol = ((size_t)b * S + s) * 2048 + (size_t)h * 128;

  float qv, kv;
  if (d < 64) {
    qv = (float)q_cont[incol + d];
    kv = (float)k_cont[incol + d];
  } else {
    const int j = d - 64;
    const int i = j & 31;
    const float inv_freq = __expf(-0.2878231366242558f * (float)i);  // 10000^(-i/32)
    const float ang = (float)s * inv_freq;
    const float c = __cosf(ang), sn = __sinf(ang);
    const float qj = (float)q_rot[incol + j];
    const float kj = (float)k_rot[incol + j];
    const float qo = (j < 32) ? -(float)q_rot[incol + j + 32] : (float)q_rot[incol + j - 32];
    const float ko = (j < 32) ? -(float)k_rot[incol + j + 32] : (float)k_rot[incol + j - 32];
    qv = qj * c + qo * sn;
    kv = kj * c + ko * sn;
  }
  const size_t o = ((size_t)(b * 16 + h) * S + s) * 128 + d;
  qh[o] = (_Float16)(qv * 0.08838834764831845f);  // 1/sqrt(128) folded into Q
  kh[o] = (_Float16)kv;
  vh[o] = (_Float16)v_full[incol + d];
}

// ---------------------------------------------------------------------------
// Causal flash attention. Block = 128 threads = 4 waves; each wave owns a full
// 16-query-row strip (softmax stats reduce within a 16-lane half). Q and K
// tiles stream via async global->LDS; V is transposed into LDS manually.
// Output written as f16 [B,S,D].
// ---------------------------------------------------------------------------
__global__ __launch_bounds__(128) void mla_flash(
    const _Float16* __restrict__ qh, const _Float16* __restrict__ kh,
    const _Float16* __restrict__ vh, _Float16* __restrict__ attn_out, int S) {
  const int QST = 136;  // 128 + 8 pad halves (272B rows, 16B aligned)
  const int VST = 72;   // 64 + 8 pad halves
  __shared__ __align__(16) _Float16 Qs[64 * 136];
  __shared__ __align__(16) _Float16 Ks[64 * 136];
  __shared__ __align__(16) _Float16 Vt[128 * 72];   // [d][key]
  __shared__ __align__(16) _Float16 Ps[4 * 16 * 72];

  const int tid  = threadIdx.x;
  const int lane = tid & 31;
  const int wave = tid >> 5;
  const int qt    = blockIdx.x;
  const int bh    = blockIdx.y;
  const int qbase = qt * 64;

  const _Float16* Qg = qh + ((size_t)bh * S + qbase) * 128;
  const _Float16* Kg = kh + (size_t)bh * S * 128;
  const _Float16* Vg = vh + (size_t)bh * S * 128;

  // Async-load Q tile (64x128 = 1024 x 16B chunks; 8 per thread)
#pragma unroll
  for (int i = 0; i < 8; ++i) {
    int c = tid + i * 128;
    int r = c >> 4, seg = c & 15;
    async_copy_b128(lds_off32(&Qs[r * QST + seg * 8]),
                    Qg + (size_t)r * 128 + seg * 8);
  }

  float m[8], l[8];
  v8f o[8];
#pragma unroll
  for (int v = 0; v < 8; ++v) { m[v] = -3.0e38f; l[v] = 0.f; }
#pragma unroll
  for (int j = 0; j < 8; ++j) o[j] = vzero8f();

  const int hs = (lane >> 4) * 8;
  const int nn = lane & 15;

  for (int kt = 0; kt <= qt; ++kt) {
    __syncthreads();  // previous tile fully consumed; Ks/Vt/Ps reusable
    // K tile: async straight copy
#pragma unroll
    for (int i = 0; i < 8; ++i) {
      int c = tid + i * 128;
      int r = c >> 4, seg = c & 15;
      async_copy_b128(lds_off32(&Ks[r * QST + seg * 8]),
                      Kg + ((size_t)kt * 64 + r) * 128 + seg * 8);
    }
    // V tile: vectorized read, transposed scatter into LDS
#pragma unroll
    for (int i = 0; i < 8; ++i) {
      int c = tid + i * 128;
      int r = c >> 4, seg = c & 15;
      v8h vv = *(const v8h*)(Vg + ((size_t)kt * 64 + r) * 128 + seg * 8);
#pragma unroll
      for (int t = 0; t < 8; ++t) Vt[(seg * 8 + t) * VST + r] = vv[t];
    }
    wait_async0();
    __syncthreads();

    // S = Q K^T : 16 rows x 64 keys; K-dim 128 = 4 WMMA steps
    v8f s0 = vzero8f(), s1 = vzero8f(), s2 = vzero8f(), s3 = vzero8f();
#pragma unroll
    for (int ks = 0; ks < 4; ++ks) {
      v16h a = load_frag_a(Qs, QST, wave * 16, ks * 32, lane);
      s0 = WMMA16(a, load_frag_b(Ks, QST, 0,  ks * 32, lane), s0);
      s1 = WMMA16(a, load_frag_b(Ks, QST, 16, ks * 32, lane), s1);
      s2 = WMMA16(a, load_frag_b(Ks, QST, 32, ks * 32, lane), s2);
      s3 = WMMA16(a, load_frag_b(Ks, QST, 48, ks * 32, lane), s3);
    }
    v8f* sacc[4] = {&s0, &s1, &s2, &s3};

    if (kt == qt) {  // causal mask on the diagonal tile
#pragma unroll
      for (int j = 0; j < 4; ++j)
#pragma unroll
        for (int v = 0; v < 8; ++v) {
          int qrow = qbase + wave * 16 + v + hs;
          int kcol = kt * 64 + j * 16 + nn;
          if (kcol > qrow) (*sacc[j])[v] = -3.0e38f;
        }
    }

    // Online softmax per row slot v
#pragma unroll
    for (int v = 0; v < 8; ++v) {
      float rm = fmaxf(fmaxf(s0[v], s1[v]), fmaxf(s2[v], s3[v]));
      for (int off = 1; off < 16; off <<= 1) rm = fmaxf(rm, __shfl_xor(rm, off, 16));
      float mnew  = fmaxf(m[v], rm);
      float scale = __expf(m[v] - mnew);
      m[v] = mnew;
      float rs = 0.f;
#pragma unroll
      for (int j = 0; j < 4; ++j) {
        float p = __expf((*sacc[j])[v] - mnew);
        (*sacc[j])[v] = p;
        rs += p;
      }
      for (int off = 1; off < 16; off <<= 1) rs += __shfl_xor(rs, off, 16);
      l[v] = l[v] * scale + rs;
#pragma unroll
      for (int j = 0; j < 8; ++j) o[j][v] *= scale;
    }

    // Stage P (16x64) to convert C-layout -> A-fragment layout
    _Float16* Pw = Ps + wave * 16 * VST;
#pragma unroll
    for (int j = 0; j < 4; ++j)
#pragma unroll
      for (int v = 0; v < 8; ++v)
        Pw[(v + hs) * VST + j * 16 + nn] = (_Float16)(*sacc[j])[v];
    __syncthreads();

    // O += P V : K-dim 64 = 2 WMMA steps; N = 128 dims = 8 tiles
#pragma unroll
    for (int ks = 0; ks < 2; ++ks) {
      v16h a = load_frag_a(Pw, VST, 0, ks * 32, lane);
#pragma unroll
      for (int j = 0; j < 8; ++j)
        o[j] = WMMA16(a, load_frag_b(Vt, VST, j * 16, ks * 32, lane), o[j]);
    }
  }

  // Write O / l as f16 into [B,S,D]
  const int b = bh >> 4, h = bh & 15;
#pragma unroll
  for (int j = 0; j < 8; ++j)
#pragma unroll
    for (int v = 0; v < 8; ++v) {
      int srow = qbase + wave * 16 + v + hs;
      int d    = j * 16 + nn;
      attn_out[((size_t)b * S + srow) * 2048 + h * 128 + d] =
          (_Float16)(o[j][v] / l[v]);
    }
}

// ---------------------------------------------------------------------------
extern "C" void kernel_launch(void* const* d_in, const int* in_sizes, int n_in,
                              void* d_out, int out_size, void* d_ws, size_t ws_size,
                              hipStream_t stream) {
  (void)in_sizes; (void)n_in; (void)out_size; (void)ws_size;
  const float* x        = (const float*)d_in[0];
  // d_in[1] = causal_mask (causality applied analytically in mla_flash)
  const float* w_lat_q  = (const float*)d_in[2];
  const float* w_lat_kv = (const float*)d_in[3];
  const float* w_rot_q  = (const float*)d_in[4];
  const float* w_rot_k  = (const float*)d_in[5];
  const float* w_q_up   = (const float*)d_in[6];
  const float* w_k_up   = (const float*)d_in[7];
  const float* w_v_up   = (const float*)d_in[8];
  const float* w_out    = (const float*)d_in[9];
  float* out = (float*)d_out;

  const int B = 2, S = 2048, D = 2048, H = 16, LQ = 1024, LKV = 512;
  const int M = B * S;  // 4096

  char* ws = (char*)d_ws;
  auto carve16 = [&](size_t n) { _Float16* p = (_Float16*)ws; ws += n * 2; return p; };
  _Float16* x16    = carve16((size_t)M * D);
  _Float16* wlq_t  = carve16((size_t)D * LQ);    // [LQ, D]
  _Float16* wlkv_t = carve16((size_t)D * LKV);   // [LKV, D]
  _Float16* wrq_t  = carve16((size_t)LQ * D);    // [D, LQ]
  _Float16* wrk_t  = carve16((size_t)LKV * D);   // [D, LKV]
  _Float16* wqu_t  = carve16((size_t)LQ * D);
  _Float16* wku_t  = carve16((size_t)LKV * D);
  _Float16* wvu_t  = carve16((size_t)LKV * D);
  _Float16* wo_t   = carve16((size_t)D * D);
  _Float16* z_q    = carve16((size_t)M * LQ);
  _Float16* z_kv   = carve16((size_t)M * LKV);
  _Float16* q_rot  = carve16((size_t)M * D);
  _Float16* q_cont = carve16((size_t)M * D);
  _Float16* k_rot  = carve16((size_t)M * D);
  _Float16* k_cont = carve16((size_t)M * D);
  _Float16* v_full = carve16((size_t)M * D);
  _Float16* qh     = carve16((size_t)M * D);
  _Float16* kh     = carve16((size_t)M * D);
  _Float16* vh     = carve16((size_t)M * D);
  _Float16* attn   = carve16((size_t)M * D);

  const dim3 blk(256);
  // One-time f16 conversion of activations + transposed weights
  cvt_f16<<<dim3(2048), blk, 0, stream>>>(x, x16, (long)M * D);
  cvt_f16_t<<<dim3(LQ / 32,  D / 8),  blk, 0, stream>>>(w_lat_q,  wlq_t,  D,  LQ);
  cvt_f16_t<<<dim3(LKV / 32, D / 8),  blk, 0, stream>>>(w_lat_kv, wlkv_t, D,  LKV);
  cvt_f16_t<<<dim3(D / 32,  LQ / 8),  blk, 0, stream>>>(w_rot_q,  wrq_t,  LQ,  D);
  cvt_f16_t<<<dim3(D / 32,  LKV / 8), blk, 0, stream>>>(w_rot_k,  wrk_t,  LKV, D);
  cvt_f16_t<<<dim3(D / 32,  LQ / 8),  blk, 0, stream>>>(w_q_up,   wqu_t,  LQ,  D);
  cvt_f16_t<<<dim3(D / 32,  LKV / 8), blk, 0, stream>>>(w_k_up,   wku_t,  LKV, D);
  cvt_f16_t<<<dim3(D / 32,  LKV / 8), blk, 0, stream>>>(w_v_up,   wvu_t,  LKV, D);
  cvt_f16_t<<<dim3(D / 32,  D / 8),   blk, 0, stream>>>(w_out,    wo_t,   D,   D);

  // Latent projections
  gemm_f16_wmma<true><<<dim3(LQ / 64,  M / 128), blk, 0, stream>>>(x16, wlq_t,  z_q,  M, LQ,  D);
  gemm_f16_wmma<true><<<dim3(LKV / 64, M / 128), blk, 0, stream>>>(x16, wlkv_t, z_kv, M, LKV, D);
  // Rotary + content up-projections
  gemm_f16_wmma<true><<<dim3(D / 64, M / 128), blk, 0, stream>>>(z_q,  wrq_t, q_rot,  M, D, LQ);
  gemm_f16_wmma<true><<<dim3(D / 64, M / 128), blk, 0, stream>>>(z_q,  wqu_t, q_cont, M, D, LQ);
  gemm_f16_wmma<true><<<dim3(D / 64, M / 128), blk, 0, stream>>>(z_kv, wrk_t, k_rot,  M, D, LKV);
  gemm_f16_wmma<true><<<dim3(D / 64, M / 128), blk, 0, stream>>>(z_kv, wku_t, k_cont, M, D, LKV);
  gemm_f16_wmma<true><<<dim3(D / 64, M / 128), blk, 0, stream>>>(z_kv, wvu_t, v_full, M, D, LKV);
  // RoPE + head assembly
  mla_assemble<<<dim3(B * S * H), dim3(128), 0, stream>>>(
      q_cont, q_rot, k_cont, k_rot, v_full, qh, kh, vh, S);
  // Causal flash attention -> attn (f16, [B,S,D])
  mla_flash<<<dim3(S / 64, B * H), dim3(128), 0, stream>>>(qh, kh, vh, attn, S);
  // Output projection (f32 out)
  gemm_f16_wmma<false><<<dim3(D / 64, M / 128), blk, 0, stream>>>(attn, wo_t, out, M, D, D);
}
